// CdfgReader_20255065768053
// MI455X (gfx1250) — compile-verified, matrix operand order checked by reading, not run on hardware
//
#include <hip/hip_runtime.h>
#include <hip/hip_bf16.h>
#include <stdint.h>

// ---------------------------------------------------------------------------
// CdfgReader GCN on MI455X (gfx1250): bf16 WMMA GEMMs + f32 accumulation.
// G=8, N=1024, F=128, H=256, L=4, B=64. All GEMM B-operands pre-transposed
// ([N][K], ld=K) so LDS staging is pure b128->b128 copies for both A and B.
// ---------------------------------------------------------------------------
constexpr int kG = 8;
constexpr int kN = 1024;
constexpr int kF = 128;
constexpr int kH = 256;
constexpr int kL = 4;
constexpr int kB = 64;

typedef __attribute__((ext_vector_type(16))) __bf16 v16bf;
typedef __attribute__((ext_vector_type(8)))  __bf16 v8bf;
typedef __attribute__((ext_vector_type(8)))  float  v8f;

__device__ __forceinline__ unsigned short f32_to_bf16(float f) {
  union { float f; uint32_t u; } v; v.f = f;
  uint32_t u = v.u;
  u += 0x7fffu + ((u >> 16) & 1u);      // round to nearest even
  return (unsigned short)(u >> 16);
}
__device__ __forceinline__ float bf16_to_f32(unsigned short h) {
  union { uint32_t u; float f; } v; v.u = ((uint32_t)h) << 16;
  return v.f;
}

// ---------------------------------------------------------------------------
// elementwise f32 -> bf16
// ---------------------------------------------------------------------------
__global__ void cvt_f32_bf16(const float* __restrict__ src,
                             unsigned short* __restrict__ dst, int n) {
  int i = blockIdx.x * blockDim.x + threadIdx.x;
  if (i < n) dst[i] = f32_to_bf16(src[i]);
}

// transpose + convert: src [K][N] f32 -> dst [N][K] bf16  (small weight matrices)
__global__ void cvt_f32_bf16_t(const float* __restrict__ src,
                               unsigned short* __restrict__ dst, int K, int N) {
  int i = blockIdx.x * blockDim.x + threadIdx.x;
  if (i < K * N) {
    int n = i / K, k = i - n * K;
    dst[i] = f32_to_bf16(src[(long long)k * N + n]);
  }
}

// ---------------------------------------------------------------------------
// Batched GEMM: C[b] (MxNc) = act( A[b](MxK, row-major) * B[b]^T + bias )
//   B operand stored TRANSPOSED: Bt (Nc x K, row-major, ld=K).
//   A base offset: graph[b]*strideA (useGraphA=1) or b*strideA (0)
//   Bt base offset: b*strideB (strideB=0 -> shared weights)
//   transC=1: bf16 output stored transposed [Nc][M] (vectorized b128 stores).
// Block: 512 threads (16 wave32). Tile: 128x128, BK=64 (2 WMMA k-steps).
// Each wave: 32x32 patch = 2x2 v_wmma_f32_16x16x32_bf16 accumulators.
// Staging values are scalar uint4s (never address-taken) so they stay in VGPRs.
// ---------------------------------------------------------------------------
#define LPITCH 72   // bf16 elements per LDS row (64 used + 8 pad); 144B rows, 16B-aligned

__launch_bounds__(512)
__global__ void gemm_bf16_wmma(
    const unsigned short* __restrict__ A, long long strideA,
    const int* __restrict__ graph, int useGraphA,
    const unsigned short* __restrict__ Bt, long long strideB,
    const float* __restrict__ bias,
    int M, int K, int Nc,
    unsigned short* __restrict__ out_bf16,
    float* __restrict__ out_f32,
    long long strideC,
    unsigned short* __restrict__ out2_bf16,
    int act,      // 0 = none, 1 = relu
    int transC)   // 1 = store bf16 output transposed [Nc][M]
{
  __shared__ __align__(16) __bf16 sA[2][128][LPITCH];
  __shared__ __align__(16) __bf16 sB[2][128][LPITCH];

  const int tid = threadIdx.x;
  const int b   = blockIdx.z;
  const long long aoff = (useGraphA ? (long long)graph[b] : (long long)b) * strideA;
  const unsigned short* __restrict__ Ag = A + aoff;
  const unsigned short* __restrict__ Bg = Bt + (long long)b * strideB;
  const int m0 = blockIdx.y * 128;
  const int n0 = blockIdx.x * 128;

  const int w    = tid >> 5;             // 0..15
  const int lane = tid & 31;
  const int wr = w & 3, wc = w >> 2;     // 4x4 wave grid
  const int mw = wr * 32, nw = wc * 32;
  const int l16 = lane & 15;
  const int klo = (lane >> 4) * 8;       // lanes 0-15: K 0..7/16..23 ; lanes 16-31: +8
  const int rowHalf = (lane >> 4) * 8;   // C/D layout: lanes 16-31 hold M = i+8

  // --- staging: by-value helpers; no allocas, values live in VGPRs
  auto gload1 = [&](const unsigned short* __restrict__ src, int rowbase, int k0,
                    int c) -> uint4 {
    int row = c >> 3, seg = c & 7;       // 8 x 16B per 128B row
    return *(const uint4*)(src + (long long)(rowbase + row) * K + k0 + seg * 8);
  };
  auto sstore1 = [&](__bf16* dst, int c, uint4 v) {
    int row = c >> 3, seg = c & 7;
    *(uint4*)(dst + row * LPITCH + seg * 8) = v;
  };
  auto loadFrag = [&](const __bf16* rowptr, int kk) -> v16bf {
    v8bf c0 = *(const v8bf*)(rowptr + kk * 32 + klo);
    v8bf c1 = *(const v8bf*)(rowptr + kk * 32 + 16 + klo);
    v16bf f;
#pragma unroll
    for (int j = 0; j < 8; ++j) { f[j] = c0[j]; f[j + 8] = c1[j]; }
    return f;
  };

  v8f acc[2][2];
#pragma unroll
  for (int r = 0; r < 2; ++r)
#pragma unroll
    for (int c = 0; c < 2; ++c) acc[r][c] = (v8f)(0.0f);

  auto computeK = [&](int cur, int kk) {
    v16bf fa[2], fb[2];
#pragma unroll
    for (int r = 0; r < 2; ++r) fa[r] = loadFrag(&sA[cur][mw + r * 16 + l16][0], kk);
#pragma unroll
    for (int c = 0; c < 2; ++c) fb[c] = loadFrag(&sB[cur][nw + c * 16 + l16][0], kk);
#pragma unroll
    for (int r = 0; r < 2; ++r)
#pragma unroll
      for (int c = 0; c < 2; ++c)
        acc[r][c] = __builtin_amdgcn_wmma_f32_16x16x32_bf16(
            false, fa[r], false, fb[c], (short)0, acc[r][c], false, false);
  };

  const int KT = K / 64;
  {
    uint4 a0 = gload1(Ag, m0, 0, tid), a1 = gload1(Ag, m0, 0, tid + 512);
    uint4 b0 = gload1(Bg, n0, 0, tid), b1 = gload1(Bg, n0, 0, tid + 512);
    sstore1(&sA[0][0][0], tid, a0); sstore1(&sA[0][0][0], tid + 512, a1);
    sstore1(&sB[0][0][0], tid, b0); sstore1(&sB[0][0][0], tid + 512, b1);
  }
  __syncthreads();

  int cur = 0;
  for (int kt = 0; kt < KT; ++kt) {
    const bool more = (kt + 1 < KT);
    uint4 ra0, ra1, rb0, rb1;         // only written+read under `more`
    if (more) {                       // issue next-tile global loads early ...
      const int k1 = (kt + 1) * 64;
      ra0 = gload1(Ag, m0, k1, tid); ra1 = gload1(Ag, m0, k1, tid + 512);
      rb0 = gload1(Bg, n0, k1, tid); rb1 = gload1(Bg, n0, k1, tid + 512);
    }
    if (kt + 2 < KT) {                // ... and global_prefetch_b8 the tile after
      __builtin_prefetch((const void*)(Ag + (long long)(m0 + (tid & 127)) * K + (kt + 2) * 64), 0, 0);
      __builtin_prefetch((const void*)(Bg + (long long)(n0 + (tid & 127)) * K + (kt + 2) * 64), 0, 0);
    }

    computeK(cur, 0);                 // ds_load_b128 + WMMA hide load latency
    if (more) {                       // commit staged tile; staging regs die here
      __bf16* dA = &sA[cur ^ 1][0][0];
      __bf16* dB = &sB[cur ^ 1][0][0];
      sstore1(dA, tid, ra0); sstore1(dA, tid + 512, ra1);
      sstore1(dB, tid, rb0); sstore1(dB, tid + 512, rb1);
    }
    computeK(cur, 1);

    __syncthreads();
    cur ^= 1;
  }

  // ---- epilogue ----
  unsigned short* __restrict__ Cb = out_bf16 ? out_bf16 + (long long)b * strideC : nullptr;
  float*          __restrict__ Cf = out_f32  ? out_f32  + (long long)b * strideC : nullptr;
  unsigned short* __restrict__ C2 = out2_bf16 ? out2_bf16 + (long long)b * strideC : nullptr;

#pragma unroll
  for (int r = 0; r < 2; ++r) {
#pragma unroll
    for (int c = 0; c < 2; ++c) {
      const int col  = n0 + nw + c * 16 + l16;
      const int rowb = m0 + mw + r * 16 + rowHalf;
      const float bv = bias ? bias[col] : 0.0f;
      if (transC && Cb) {
        // transposed store: lane's 8 rows are contiguous -> one b128 store
        union { unsigned short u16[8]; uint4 v; } pk;
#pragma unroll
        for (int i = 0; i < 8; ++i) {
          float v = acc[r][c][i] + bv;
          if (act == 1) v = v > 0.0f ? v : 0.0f;
          pk.u16[i] = f32_to_bf16(v);
        }
        *(uint4*)(Cb + (long long)col * M + rowb) = pk.v;
      } else {
#pragma unroll
        for (int i = 0; i < 8; ++i) {
          float v = acc[r][c][i] + bv;
          if (act == 1) v = v > 0.0f ? v : 0.0f;
          const long long idx = (long long)(rowb + i) * Nc + col;
          if (Cf) Cf[idx] = v;
          if (Cb) Cb[idx] = f32_to_bf16(v);
          if (C2) C2[idx] = f32_to_bf16(v);
        }
      }
    }
  }
}

// ---------------------------------------------------------------------------
// in-place softmax over H=256 per (b,n) row
// ---------------------------------------------------------------------------
__global__ void softmax_rows(float* __restrict__ h) {
  __shared__ float red[256];
  const int t = threadIdx.x;
  float* p = h + (long long)blockIdx.x * kH;
  float v = p[t];
  red[t] = v; __syncthreads();
  for (int s = 128; s > 0; s >>= 1) { if (t < s) red[t] = fmaxf(red[t], red[t + s]); __syncthreads(); }
  const float mx = red[0]; __syncthreads();
  const float e = __expf(v - mx);
  red[t] = e; __syncthreads();
  for (int s = 128; s > 0; s >>= 1) { if (t < s) red[t] += red[t + s]; __syncthreads(); }
  p[t] = e / red[0];
}

// ---------------------------------------------------------------------------
// out[b,h] = sum_n mask[b,n] * (softmax_x[b,n,h] + to_add[b,n,h]) / max(cnt,1)
// ---------------------------------------------------------------------------
__global__ void masked_mean(const float* __restrict__ p,
                            const unsigned short* __restrict__ toadd,
                            const unsigned char* __restrict__ mask,
                            float* __restrict__ out) {
  const int b = blockIdx.x, t = threadIdx.x;
  const float* pb = p + (long long)b * kN * kH;
  const unsigned short* ab = toadd + (long long)b * kN * kH;
  const unsigned char* mb = mask + (long long)b * kN;
  float acc = 0.0f, cnt = 0.0f;
  for (int n = 0; n < kN; ++n) {
    if (mb[n]) {
      acc += pb[(long long)n * kH + t] + bf16_to_f32(ab[(long long)n * kH + t]);
      cnt += 1.0f;
    }
  }
  if (cnt < 1.0f) cnt = 1.0f;
  out[b * kH + t] = acc / cnt;
}

// ---------------------------------------------------------------------------
// launch
// ---------------------------------------------------------------------------
extern "C" void kernel_launch(void* const* d_in, const int* in_sizes, int n_in,
                              void* d_out, int out_size, void* d_ws, size_t ws_size,
                              hipStream_t stream) {
  const int*            graph = (const int*)d_in[0];
  const unsigned char*  mask  = (const unsigned char*)d_in[1];  // jnp bool_ = 1 byte
  const float*          xs    = (const float*)d_in[2];          // [G,N,F]
  const float*          as    = (const float*)d_in[3];          // [G,N,N]
  const float*          Win   = (const float*)d_in[4];          // [F,H]
  const float*          b_in  = (const float*)d_in[5];          // [H]
  const float*          Wsp   = (const float*)d_in[6];          // [L,H,H]
  const float*          bsp   = (const float*)d_in[7];          // [L,H]
  float* out = (float*)d_out;                                   // [B,H] f32

  // workspace layout (256B-aligned slabs), ~187 MB total
  char* ws = (char*)d_ws;
  size_t off = 0;
  auto alloc = [&](size_t bytes) -> char* {
    char* p = ws + off;
    off = (off + bytes + 255) & ~(size_t)255;
    return p;
  };
  unsigned short* xs_bf  = (unsigned short*)alloc((size_t)kG * kN * kF * 2);  //   2 MB
  unsigned short* as_bf  = (unsigned short*)alloc((size_t)kG * kN * kN * 2);  //  16.8 MB
  unsigned short* winT   = (unsigned short*)alloc((size_t)kF * kH * 2);       //  [H][F]
  unsigned short* wsT    = (unsigned short*)alloc((size_t)kL * kH * kH * 2);  //  [L][H][H] transposed
  unsigned short* xbuf   = (unsigned short*)alloc((size_t)kB * kN * kH * 2);  //  33.5 MB  [N][H]
  unsigned short* ybufT  = (unsigned short*)alloc((size_t)kB * kN * kH * 2);  //  33.5 MB  [H][N] (y^T)
  unsigned short* toadd  = (unsigned short*)alloc((size_t)kB * kN * kH * 2);  //  33.5 MB
  float*          hbuf   = (float*)alloc((size_t)kB * kN * kH * 4);           //  67.1 MB

  auto cvt = [&](const float* s, unsigned short* d, int n) {
    cvt_f32_bf16<<<(n + 255) / 256, 256, 0, stream>>>(s, d, n);
  };
  cvt(xs, xs_bf, kG * kN * kF);
  cvt(as, as_bf, kG * kN * kN);
  // weights converted AND transposed to [N][K]
  cvt_f32_bf16_t<<<(kF * kH + 255) / 256, 256, 0, stream>>>(Win, winT, kF, kH);
  for (int i = 0; i < kL; ++i)
    cvt_f32_bf16_t<<<(kH * kH + 255) / 256, 256, 0, stream>>>(
        Wsp + (size_t)i * kH * kH, wsT + (size_t)i * kH * kH, kH, kH);

  const dim3 blk(512);
  const dim3 grd(kH / 128, kN / 128, kB);   // (Nc/128, M/128, B)
  const long long sC = (long long)kN * kH;

  // input dense: x = relu(xs[g] @ W_in + b_in) -> xbuf (bf16) and to_add copy
  gemm_bf16_wmma<<<grd, blk, 0, stream>>>(
      xs_bf, (long long)kN * kF, graph, 1,
      winT, 0, b_in,
      kN, kF, kH,
      xbuf, nullptr, sC, toadd, /*act=*/1, /*transC=*/0);

  for (int i = 0; i < kL; ++i) {
    // y^T = (x @ Ws[i])^T    (per-sample A, shared Bt, transposed bf16 output)
    gemm_bf16_wmma<<<grd, blk, 0, stream>>>(
        xbuf, sC, graph, 0,
        wsT + (size_t)i * kH * kH, 0, nullptr,
        kN, kH, kH,
        ybufT, nullptr, sC, nullptr, /*act=*/0, /*transC=*/1);
    // h = As[g] @ y + bs[i]  (graph-indexed A; Bt = y^T per sample, ld=N)
    if (i < kL - 1) {
      gemm_bf16_wmma<<<grd, blk, 0, stream>>>(
          as_bf, (long long)kN * kN, graph, 1,
          ybufT, sC, bsp + (size_t)i * kH,
          kN, kN, kH,
          xbuf, nullptr, sC, nullptr, /*act=*/1, /*transC=*/0);
    } else {
      gemm_bf16_wmma<<<grd, blk, 0, stream>>>(
          as_bf, (long long)kN * kN, graph, 1,
          ybufT, sC, bsp + (size_t)i * kH,
          kN, kN, kH,
          nullptr, hbuf, sC, nullptr, /*act=*/0, /*transC=*/0);
    }
  }

  softmax_rows<<<kB * kN, 256, 0, stream>>>(hbuf);
  masked_mean<<<kB, 256, 0, stream>>>(hbuf, toadd, mask, out);
}
